// SelfMatching_52974126629614
// MI455X (gfx1250) — matrix operand order repaired and checked by reading, not compiled, not run on hardware
//
#include <hip/hip_runtime.h>
#include <stdint.h>

// ---------------------------------------------------------------------------
// Problem constants (B=64, T=512, D=256)
// ---------------------------------------------------------------------------
static constexpr int CB = 64;
static constexpr int CT = 512;
static constexpr int CD = 256;

// ---------------------------------------------------------------------------
// bf16 WMMA plumbing (CDNA5 / gfx1250, wave32)
// ---------------------------------------------------------------------------
typedef __attribute__((ext_vector_type(16))) __bf16 v16bf;
typedef __attribute__((ext_vector_type(8)))  float  v8f;

union FragU {
  v16bf v;
  uint4 q[2];
  unsigned short s[16];
  __bf16 b[16];
};

// Native bf16 convert (backend selects v_cvt hardware op; RNE either way)
__device__ __forceinline__ unsigned short f2bf(float x) {
  union { __bf16 b; unsigned short s; } c;
  c.b = (__bf16)x;
  return c.s;
}
__device__ __forceinline__ float bf2f(unsigned short h) {
  union { unsigned short s; __bf16 b; } c;
  c.s = h;
  return (float)c.b;
}
__device__ __forceinline__ float sigm(float x) { return 1.0f / (1.0f + __expf(-x)); }
__device__ __forceinline__ v8f zero8() { v8f z = {0.f,0.f,0.f,0.f,0.f,0.f,0.f,0.f}; return z; }

#if __has_builtin(__builtin_amdgcn_tanhf)
#define dev_tanh(x) __builtin_amdgcn_tanhf(x)
#elif __has_builtin(__builtin_amdgcn_tanh_f32)
#define dev_tanh(x) __builtin_amdgcn_tanh_f32(x)
#else
#define dev_tanh(x) tanhf(x)
#endif

__device__ __forceinline__ v8f wmma_bf16(v16bf a, v16bf b, v8f c) {
  return __builtin_amdgcn_wmma_f32_16x16x32_bf16(false, a, false, b, (short)0, c,
                                                 false, false);
}

// ---------------------------------------------------------------------------
// Async global->LDS copy (CDNA5 GLOBAL_LOAD_ASYNC_TO_LDS_B128, ASYNCcnt).
// Probe result: builtin's first param is `int __vector(4) __device__*`, i.e.
// a 16-byte int vector pointer in AS1; LDS side is the AS3 equivalent.
// ---------------------------------------------------------------------------
#if __has_builtin(__builtin_amdgcn_global_load_async_to_lds_b128)
#define HAVE_ASYNC_LDS 1
#endif

typedef int v4i_vs __attribute__((vector_size(16)));
typedef __attribute__((address_space(1))) v4i_vs* gptr_b128;
typedef __attribute__((address_space(3))) v4i_vs* lptr_b128;

__device__ __forceinline__ void async_copy_b128(const unsigned short* g,
                                                unsigned short* l) {
#ifdef HAVE_ASYNC_LDS
  __builtin_amdgcn_global_load_async_to_lds_b128((gptr_b128)(g), (lptr_b128)(l),
                                                 0, 0);
#else
  *(uint4*)l = *(const uint4*)g;
#endif
}
__device__ __forceinline__ void async_join() {
#ifdef HAVE_ASYNC_LDS
#if __has_builtin(__builtin_amdgcn_s_wait_asynccnt)
  __builtin_amdgcn_s_wait_asynccnt(0);
#else
  asm volatile("s_wait_asynccnt 0x0" ::: "memory");
#endif
#endif
}

// A fragment (16x32 bf16) from a row-major ushort array (global or LDS).
// ISA layout: lanes 0-15 hold row r, K = k0+0..7 (v0..3) and k0+16..23 (v4..7);
// lanes 16-31 hold K = k0+8..15 and k0+24..31.  Two b128 loads per lane.
__device__ __forceinline__ v16bf load_fragA(const unsigned short* p, int ld,
                                            int row0, int k0) {
  const int lane = threadIdx.x & 31;
  const int r = lane & 15, lh = lane >> 4;
  const unsigned short* a = p + (size_t)(row0 + r) * ld + k0 + lh * 8;
  FragU f;
  f.q[0] = *(const uint4*)(a);
  f.q[1] = *(const uint4*)(a + 16);
  return f.v;
}

// B fragment (32x16 bf16) from an N-major (pre-transposed) array p[n*ld+k].
// Lane l holds column n0+(l&15); k = k0 + (l>>4)*8 + 0..7 (v0..3) and +16..23 (v4..7).
__device__ __forceinline__ v16bf load_fragB(const unsigned short* p, int ld,
                                            int n0, int k0) {
  const int lane = threadIdx.x & 31;
  const int n = lane & 15, lh = lane >> 4;
  const unsigned short* a = p + (size_t)(n0 + n) * ld + k0 + lh * 8;
  FragU f;
  f.q[0] = *(const uint4*)(a);
  f.q[1] = *(const uint4*)(a + 16);
  return f.v;
}

// A fragment built on the fly from fp32 LDS (GRU scan: A = h).
__device__ __forceinline__ v16bf cvt_fragA_f32(const float* base, int ld,
                                               int row0, int k0) {
  const int lane = threadIdx.x & 31;
  const int r = lane & 15, lh = lane >> 4;
  const float* p = base + (row0 + r) * ld + k0 + lh * 8;
  float4 a0 = *(const float4*)(p);
  float4 a1 = *(const float4*)(p + 4);
  float4 a2 = *(const float4*)(p + 16);
  float4 a3 = *(const float4*)(p + 20);
  FragU f;
  f.b[0]=(__bf16)a0.x;  f.b[1]=(__bf16)a0.y;  f.b[2]=(__bf16)a0.z;  f.b[3]=(__bf16)a0.w;
  f.b[4]=(__bf16)a1.x;  f.b[5]=(__bf16)a1.y;  f.b[6]=(__bf16)a1.z;  f.b[7]=(__bf16)a1.w;
  f.b[8]=(__bf16)a2.x;  f.b[9]=(__bf16)a2.y;  f.b[10]=(__bf16)a2.z; f.b[11]=(__bf16)a2.w;
  f.b[12]=(__bf16)a3.x; f.b[13]=(__bf16)a3.y; f.b[14]=(__bf16)a3.z; f.b[15]=(__bf16)a3.w;
  return f.v;
}

// ---------------------------------------------------------------------------
// Kernel 1: passage (fp32) -> pa_bf[:, 0:D] (bf16).  8 elements per thread.
// ---------------------------------------------------------------------------
__global__ __launch_bounds__(256) void copy_passage_kernel(
    const float* __restrict__ passage, unsigned short* __restrict__ pa_bf) {
  int gid = blockIdx.x * 256 + threadIdx.x;   // over B*T*D/8
  int row = gid >> 5;                          // 32 octets per row
  int dq  = (gid & 31) << 3;
  const float* src = passage + (size_t)row * CD + dq;
  float4 f0 = *(const float4*)(src);
  float4 f1 = *(const float4*)(src + 4);
  union { uint4 q; __bf16 b[8]; } u;
  u.b[0]=(__bf16)f0.x; u.b[1]=(__bf16)f0.y; u.b[2]=(__bf16)f0.z; u.b[3]=(__bf16)f0.w;
  u.b[4]=(__bf16)f1.x; u.b[5]=(__bf16)f1.y; u.b[6]=(__bf16)f1.z; u.b[7]=(__bf16)f1.w;
  *(uint4*)(pa_bf + (size_t)row * (2 * CD) + dq) = u.q;
}

// ---------------------------------------------------------------------------
// Kernel 2: transpose + bf16-convert weight [K,N] -> [N][K] for B fragments
// ---------------------------------------------------------------------------
__global__ __launch_bounds__(256) void transpose_bf16_kernel(
    const float* __restrict__ src, unsigned short* __restrict__ dst,
    int K, int N) {
  int idx = blockIdx.x * 256 + threadIdx.x;
  if (idx >= K * N) return;
  int n = idx % N, k = idx / N;                    // coalesced read over n
  dst[(size_t)n * K + k] = f2bf(src[idx]);
}

// ---------------------------------------------------------------------------
// Kernel 3: generic bf16-WMMA GEMM, zero LDS, zero barriers.
//   A  : bf16 [M,K] row-major (L2-resident), Bt : bf16 [N,K] pre-transposed.
//   mode 0: Cf = acc            (fp32 out)
//   mode 1: Cf = acc + bias[n]  (fp32 out)
//   mode 2: Ch = bf16(sigmoid(acc) * gate[row,col])   (gated projection)
//   mode 3: Ch = bf16(acc)
// Tile 128x128/WG, 8 waves each 32x64 (2x4 WMMA tiles), BK=32.
// ---------------------------------------------------------------------------
__global__ __launch_bounds__(256) void gemm_bf16_kernel(
    const unsigned short* __restrict__ A, int lda,
    const unsigned short* __restrict__ Bt, int ldb,
    const float* __restrict__ bias,
    const unsigned short* __restrict__ gate, int ldg,
    float* __restrict__ Cf, unsigned short* __restrict__ Ch, int ldc,
    int K, int mode) {
  const int m0 = blockIdx.y * 128;
  const int n0 = blockIdx.x * 128;
  const int tid = threadIdx.x;
  const int w  = tid >> 5;
  const int wy = w & 3;        // 4 row bands of 32
  const int wx = w >> 2;       // 2 col bands of 64

  v8f acc[2][4];
#pragma unroll
  for (int i = 0; i < 2; ++i)
#pragma unroll
    for (int j = 0; j < 4; ++j) acc[i][j] = zero8();

  for (int k0 = 0; k0 < K; k0 += 32) {
    v16bf af0 = load_fragA(A, lda, m0 + wy * 32 + 0, k0);
    v16bf af1 = load_fragA(A, lda, m0 + wy * 32 + 16, k0);
#pragma unroll
    for (int nt = 0; nt < 4; ++nt) {
      v16bf bfr = load_fragB(Bt, ldb, n0 + wx * 64 + nt * 16, k0);
      acc[0][nt] = wmma_bf16(af0, bfr, acc[0][nt]);
      acc[1][nt] = wmma_bf16(af1, bfr, acc[1][nt]);
    }
  }

  const int lane = tid & 31, lh = lane >> 4, nn = lane & 15;
#pragma unroll
  for (int mt = 0; mt < 2; ++mt)
#pragma unroll
    for (int nt = 0; nt < 4; ++nt) {
      int col = n0 + wx * 64 + nt * 16 + nn;
      float bv = bias ? bias[col] : 0.0f;
#pragma unroll
      for (int r = 0; r < 8; ++r) {
        int row = m0 + wy * 32 + mt * 16 + r + 8 * lh;
        float v = acc[mt][nt][r] + bv;
        if (mode <= 1) {
          Cf[(size_t)row * ldc + col] = v;
        } else if (mode == 2) {
          float gv = bf2f(gate[(size_t)row * ldg + col]);
          Ch[(size_t)row * ldc + col] = f2bf(sigm(v) * gv);
        } else {
          Ch[(size_t)row * ldc + col] = f2bf(v);
        }
      }
    }
}

// ---------------------------------------------------------------------------
// Kernel 4: attention pre-pass.  One workgroup per t (grid=512, 256 thr).
// WvT + bq_t pulled into LDS with async global->LDS b128 copies (ASYNCcnt).
// For j=0..511: s_j[64,256] = tanh(ap[:,j,:]+bq[:,t,:]) @ Wv (bf16 WMMA),
// softmax over b in-workgroup, c accumulated, written bf16 to pa_bf[:,t,D:2D].
// LDS: WvT bf16 [256][264] | bq_t bf16 [64][264] | sbuf f32 [64][257] |
//      cbuf f32 [64][257]   (~294 KB of the WGP's 320 KB)
// ---------------------------------------------------------------------------
static constexpr int WVT_LD = 264;   // bf16 elements, %8==0 for b128 loads
static constexpr int SB_LD  = 257;   // f32, ≡1 mod 64 banks: conflict-free cols
static constexpr size_t ATT_SMEM =
    (size_t)(256 * WVT_LD + 64 * WVT_LD) * 2 + (size_t)2 * 64 * SB_LD * 4;

__global__ __launch_bounds__(256) void attention_kernel(
    const float* __restrict__ ap, const unsigned short* __restrict__ bq_bf,
    const float* __restrict__ passage, const unsigned short* __restrict__ WvT_g,
    unsigned short* __restrict__ pa_bf) {
  extern __shared__ unsigned char smem[];
  unsigned short* WvT = (unsigned short*)smem;                     // [256][264]
  unsigned short* bqs = WvT + 256 * WVT_LD;                        // [64][264]
  float* sbuf = (float*)(smem + (size_t)(256 + 64) * WVT_LD * 2);  // [64][257]
  float* cbuf = sbuf + 64 * SB_LD;                                 // [64][257]

  const int t   = blockIdx.x;
  const int tid = threadIdx.x;

  // async preload Wv^T (bf16, reused across all 512 j iterations)
  for (int idx = tid; idx < 256 * 32; idx += 256) {
    int n = idx >> 5, seg = (idx & 31) * 8;
    async_copy_b128(WvT_g + (size_t)n * 256 + seg, WvT + n * WVT_LD + seg);
  }
  // async preload bq[:,t,:] (bf16)
  for (int idx = tid; idx < 64 * 32; idx += 256) {
    int b = idx >> 5, seg = (idx & 31) * 8;
    async_copy_b128(bq_bf + ((size_t)(b * CT + t)) * CD + seg,
                    bqs + b * WVT_LD + seg);
  }
  for (int i = tid; i < 64 * SB_LD; i += 256) cbuf[i] = 0.0f;
  async_join();
  __syncthreads();

  const int w     = tid >> 5;
  const int mt    = w & 3;              // M tile (16 rows of b)
  const int m0    = mt * 16;
  const int nbase = (w >> 2) * 128;     // 8 N tiles per wave
  const int lane  = tid & 31;
  const int lh    = lane >> 4;
  const int nn    = lane & 15;
  const int brow  = m0 + (lane & 15);   // this lane's A row (batch index)

  for (int j = 0; j < CT; ++j) {
    v8f acc[8];
#pragma unroll
    for (int i = 0; i < 8; ++i) acc[i] = zero8();

#pragma unroll
    for (int kk = 0; kk < 8; ++kk) {
      const int k0 = kk * 32;
      // --- build A fragment: tanh(ap + bq) -> bf16 (ap fp32 from L2)
      const float* gap = ap + ((size_t)(brow * CT + j)) * CD + k0 + lh * 8;
      const unsigned short* gbq = bqs + brow * WVT_LD + k0 + lh * 8;
      float4 a0 = *(const float4*)(gap);
      float4 a1 = *(const float4*)(gap + 4);
      float4 a2 = *(const float4*)(gap + 16);
      float4 a3 = *(const float4*)(gap + 20);
      FragU qb;
      qb.q[0] = *(const uint4*)(gbq);
      qb.q[1] = *(const uint4*)(gbq + 16);
      float av[16] = {a0.x,a0.y,a0.z,a0.w, a1.x,a1.y,a1.z,a1.w,
                      a2.x,a2.y,a2.z,a2.w, a3.x,a3.y,a3.z,a3.w};
      FragU fa;
#pragma unroll
      for (int i = 0; i < 16; ++i)
        fa.b[i] = (__bf16)dev_tanh(av[i] + bf2f(qb.s[i]));
      // --- 8 WMMA against resident WvT
#pragma unroll
      for (int nt = 0; nt < 8; ++nt) {
        v16bf bfr = load_fragB(WvT, WVT_LD, nbase + nt * 16, k0);
        acc[nt] = wmma_bf16(fa.v, bfr, acc[nt]);
      }
    }

    // --- scatter s tiles to LDS (padded stride: column reads conflict-free)
#pragma unroll
    for (int nt = 0; nt < 8; ++nt)
#pragma unroll
      for (int r = 0; r < 8; ++r)
        sbuf[(m0 + r + 8 * lh) * SB_LD + nbase + nt * 16 + nn] = acc[nt][r];
    __syncthreads();

    // --- softmax over b (64 values) per column d = tid, then accumulate c
    {
      const int d = tid;
      float mx = -3.0e38f;
      for (int b2 = 0; b2 < 64; ++b2) mx = fmaxf(mx, sbuf[b2 * SB_LD + d]);
      float sum = 0.0f;
      for (int b2 = 0; b2 < 64; ++b2) sum += __expf(sbuf[b2 * SB_LD + d] - mx);
      float inv = 1.0f / sum;
      for (int b2 = 0; b2 < 64; ++b2) {
        float a = __expf(sbuf[b2 * SB_LD + d] - mx) * inv;
        cbuf[b2 * SB_LD + d] +=
            a * passage[((size_t)(b2 * CT + j)) * CD + d];
      }
    }
    __syncthreads();
  }

  // --- write c_t (bf16) into pa_bf[:, t, D:2D]
  for (int b2 = 0; b2 < 64; ++b2)
    pa_bf[((size_t)(b2 * CT + t)) * (2 * CD) + CD + tid] =
        f2bf(cbuf[b2 * SB_LD + tid]);
}

// ---------------------------------------------------------------------------
// Kernel 5: sequential GRU scan (both cells scanned forward, faithful to the
// reference).  Single workgroup, 1024 threads / 32 waves: waves 0-15 -> h_f
// GEMM, 16-31 -> h_b.  h kept in LDS fp32; gh = h@Whh + bhh on WMMA per step.
// ---------------------------------------------------------------------------
static constexpr int H_LD = 264;   // fp32 stride, %8==0 for b128 loads
static constexpr size_t SCAN_SMEM = (size_t)2 * 64 * H_LD * 4;

__global__ __launch_bounds__(1024) void scan_kernel(
    const float* __restrict__ gi_f, const float* __restrict__ gi_b,
    const unsigned short* __restrict__ WhhT_f,
    const unsigned short* __restrict__ WhhT_b,
    const float* __restrict__ bhh_f, const float* __restrict__ bhh_b,
    float* __restrict__ gh_ws, float* __restrict__ out) {
  extern __shared__ unsigned char smem[];
  float* hmem = (float*)smem;                       // [2][64][264]
  const int tid = threadIdx.x;

  for (int i = tid; i < 2 * 64 * H_LD; i += 1024) hmem[i] = 0.0f;
  __syncthreads();

  const int w  = tid >> 5;
  const int g  = w >> 4;                 // 0 = forward cell, 1 = backward cell
  const int wl = w & 15;
  const int mt = wl & 3;                 // M tile (rows of b)
  const int nbase = (wl >> 2) * 12;      // 12 N tiles per wave (2 passes of 6)
  const unsigned short* WT = g ? WhhT_b : WhhT_f;
  const float* bh = g ? bhh_b : bhh_f;
  float* hL = hmem + g * 64 * H_LD;
  const int lane = tid & 31, lh = lane >> 4, nn = lane & 15;

  for (int t = 0; t < CT; ++t) {
    // ---- gh = h @ Whh (N-major bf16) + bhh   (two passes of 6 tiles)
#pragma unroll
    for (int np = 0; np < 2; ++np) {
      v8f acc[6];
#pragma unroll
      for (int i = 0; i < 6; ++i) acc[i] = zero8();
#pragma unroll
      for (int kk = 0; kk < 8; ++kk) {
        const int k0 = kk * 32;
        v16bf fa = cvt_fragA_f32(hL, H_LD, mt * 16, k0);
#pragma unroll
        for (int nt = 0; nt < 6; ++nt) {
          v16bf bfr = load_fragB(WT, 256, (nbase + np * 6 + nt) * 16, k0);
          acc[nt] = wmma_bf16(fa, bfr, acc[nt]);
        }
      }
#pragma unroll
      for (int nt = 0; nt < 6; ++nt) {
#pragma unroll
        for (int r = 0; r < 8; ++r) {
          int col = (nbase + np * 6 + nt) * 16 + nn;
          int row = mt * 16 + r + 8 * lh;
          gh_ws[(size_t)g * 64 * 768 + (size_t)row * 768 + col] =
              acc[nt][r] + bh[col];
        }
      }
    }
    __threadfence();
    __syncthreads();

    // ---- gate math + h update + output write
    {
      const int gg  = tid >> 9;
      const int rem = tid & 511;
      const int d   = rem & 255;
      const int b0  = (rem >> 8) * 32;
      const float* gi = gg ? gi_b : gi_f;
      float* hU = hmem + gg * 64 * H_LD;
      for (int b2 = b0; b2 < b0 + 32; ++b2) {
        size_t gio = ((size_t)(b2 * CT + t)) * 768;
        float ir = gi[gio + d];
        float iz = gi[gio + 256 + d];
        float in_ = gi[gio + 512 + d];
        const float* ghp = gh_ws + (size_t)gg * 64 * 768 + (size_t)b2 * 768;
        float hr = ghp[d], hz = ghp[256 + d], hn = ghp[512 + d];
        float ho = hU[b2 * H_LD + d];
        float r = sigm(ir + hr);
        float z = sigm(iz + hz);
        float nv = dev_tanh(in_ + r * hn);
        float hnew = (1.0f - z) * nv + z * ho;
        hU[b2 * H_LD + d] = hnew;
        out[((size_t)(b2 * CT + t)) * (2 * CD) + gg * CD + d] = hnew;
      }
    }
    __syncthreads();
  }
}

// ---------------------------------------------------------------------------
// Host-side launch.  Workspace layout (bytes):
//   ap     [B,T,D]  f32     33,554,432   (kept fp32: feeds tanh)
//   bq_bf  [B,T,D]  bf16    16,777,216
//   pa_bf  [B,T,2D] bf16    33,554,432
//   x_bf   [B,T,2D] bf16    33,554,432
//   gi_f   [B,T,3D] f32    100,663,296
//   gi_b   [B,T,3D] f32    100,663,296
//   gh_ws  [2,64,3D] f32       393,216
//   transposed bf16 weights      ~3.3 MB      (total ~322 MB)
// ---------------------------------------------------------------------------
extern "C" void kernel_launch(void* const* d_in, const int* in_sizes, int n_in,
                              void* d_out, int out_size, void* d_ws,
                              size_t ws_size, hipStream_t stream) {
  (void)in_sizes; (void)n_in; (void)out_size; (void)ws_size;
  const float* passage = (const float*)d_in[0];
  const float* Wvp1    = (const float*)d_in[1];
  const float* Wvp2    = (const float*)d_in[2];
  const float* Wv      = (const float*)d_in[3];
  const float* Wg      = (const float*)d_in[4];
  const float* Wih_f   = (const float*)d_in[5];
  const float* Whh_f   = (const float*)d_in[6];
  const float* bih_f   = (const float*)d_in[7];
  const float* bhh_f   = (const float*)d_in[8];
  const float* Wih_b   = (const float*)d_in[9];
  const float* Whh_b   = (const float*)d_in[10];
  const float* bih_b   = (const float*)d_in[11];
  const float* bhh_b   = (const float*)d_in[12];
  float* out = (float*)d_out;

  char* ws = (char*)d_ws;
  float*          ap     = (float*)(ws);
  unsigned short* bq_bf  = (unsigned short*)(ws + 33554432ull);
  unsigned short* pa_bf  = (unsigned short*)(ws + 50331648ull);
  unsigned short* x_bf   = (unsigned short*)(ws + 83886080ull);
  float*          gi_f   = (float*)(ws + 117440512ull);
  float*          gi_b   = (float*)(ws + 218103808ull);
  float*          gh     = (float*)(ws + 318767104ull);
  unsigned short* Wvp1T  = (unsigned short*)(ws + 319160320ull);
  unsigned short* Wvp2T  = (unsigned short*)(ws + 319291392ull);
  unsigned short* WvT    = (unsigned short*)(ws + 319422464ull);
  unsigned short* WgT    = (unsigned short*)(ws + 319553536ull);
  unsigned short* WihT_f = (unsigned short*)(ws + 320077824ull);
  unsigned short* WihT_b = (unsigned short*)(ws + 320864256ull);
  unsigned short* WhhT_f = (unsigned short*)(ws + 321650688ull);
  unsigned short* WhhT_b = (unsigned short*)(ws + 322043904ull);

  const int MR = CB * CT;  // 32768 flattened (b,t) rows

  // 1) bf16 operand prep (all L2-resident afterwards)
  copy_passage_kernel<<<(MR * CD / 8) / 256, 256, 0, stream>>>(passage, pa_bf);
  transpose_bf16_kernel<<<256, 256, 0, stream>>>(Wvp1, Wvp1T, CD, CD);
  transpose_bf16_kernel<<<256, 256, 0, stream>>>(Wvp2, Wvp2T, CD, CD);
  transpose_bf16_kernel<<<256, 256, 0, stream>>>(Wv, WvT, CD, CD);
  transpose_bf16_kernel<<<1024, 256, 0, stream>>>(Wg, WgT, 2 * CD, 2 * CD);
  transpose_bf16_kernel<<<1536, 256, 0, stream>>>(Wih_f, WihT_f, 2 * CD, 3 * CD);
  transpose_bf16_kernel<<<1536, 256, 0, stream>>>(Wih_b, WihT_b, 2 * CD, 3 * CD);
  transpose_bf16_kernel<<<768, 256, 0, stream>>>(Whh_f, WhhT_f, CD, 3 * CD);
  transpose_bf16_kernel<<<768, 256, 0, stream>>>(Whh_b, WhhT_b, CD, 3 * CD);

  // 2) ap = passage @ Wvp1 (fp32 out) ; bq = passage @ Wvp2 (bf16 out)
  gemm_bf16_kernel<<<dim3(CD / 128, MR / 128), 256, 0, stream>>>(
      pa_bf, 2 * CD, Wvp1T, CD, nullptr, nullptr, 0, ap, nullptr, CD, CD, 0);
  gemm_bf16_kernel<<<dim3(CD / 128, MR / 128), 256, 0, stream>>>(
      pa_bf, 2 * CD, Wvp2T, CD, nullptr, nullptr, 0, nullptr, bq_bf, CD, CD, 3);

  // 3) attention pre-pass: c_t for every t in parallel -> pa_bf[:, :, D:2D]
  attention_kernel<<<CT, 256, ATT_SMEM, stream>>>(ap, bq_bf, passage, WvT, pa_bf);

  // 4) x = sigmoid(pa @ Wg) * pa  (bf16 out)
  gemm_bf16_kernel<<<dim3((2 * CD) / 128, MR / 128), 256, 0, stream>>>(
      pa_bf, 2 * CD, WgT, 2 * CD, nullptr, pa_bf, 2 * CD, nullptr, x_bf,
      2 * CD, 2 * CD, 2);

  // 5) gi = x @ Wih + bih for both cells (fp32 out)
  gemm_bf16_kernel<<<dim3((3 * CD) / 128, MR / 128), 256, 0, stream>>>(
      x_bf, 2 * CD, WihT_f, 2 * CD, bih_f, nullptr, 0, gi_f, nullptr, 3 * CD,
      2 * CD, 1);
  gemm_bf16_kernel<<<dim3((3 * CD) / 128, MR / 128), 256, 0, stream>>>(
      x_bf, 2 * CD, WihT_b, 2 * CD, bih_b, nullptr, 0, gi_b, nullptr, 3 * CD,
      2 * CD, 1);

  // 6) sequential scan (single workgroup, WMMA recurrent GEMMs)
  scan_kernel<<<1, 1024, SCAN_SMEM, stream>>>(gi_f, gi_b, WhhT_f, WhhT_b,
                                              bhh_f, bhh_b, gh, out);
}
